// _GroupMambaRuntime_73632919322670
// MI455X (gfx1250) — compile-verified, matrix operand order checked by reading, not run on hardware
//
#include <hip/hip_runtime.h>
#include <hip/hip_bf16.h>

// Problem dims (fixed by reference)
#define B_ 8
#define S_ 4096
#define H_ 1024
#define R_ 128
#define N_ 256
#define BS_ (B_ * S_)          // 32768 rows for all GEMMs
#define NCHUNK 32
#define CSIZE (S_ / NCHUNK)    // 128

typedef __bf16 bf16;
typedef __attribute__((ext_vector_type(16))) __bf16 v16bf;
typedef __attribute__((ext_vector_type(8)))  float  v8f;

#define WMMA_BF16(a, b, c) \
  __builtin_amdgcn_wmma_f32_16x16x32_bf16(false, (a), false, (b), (short)0, (c), false, false)

// ---------------------------------------------------------------------------
// Fragment loaders. CDNA5 v_wmma_f32_16x16x32_bf16 layout (ISA 7.12.2):
//   A 16x32: lane L<16 holds row L, elems {K0..7, K16..23}; lane L+16 holds
//   row L, elems {K8..15, K24..31}. B (32x16) comes from W row-major (since
//   out = X*W^T => B[k][n] = W[n][k]) with identical per-lane striping, with
//   "row" = output column n. Pointer passed = base + row*ld + k0 + half*8.
// ---------------------------------------------------------------------------
__device__ __forceinline__ v16bf frag_from_f32(const float* __restrict__ p) {
  v16bf f;
#pragma unroll
  for (int i = 0; i < 8; ++i) {
    f[i]     = (__bf16)p[i];        // K = k0 + half*8 + i
    f[i + 8] = (__bf16)p[i + 16];   // K = k0 + 16 + half*8 + i
  }
  return f;
}

__device__ __forceinline__ v16bf frag_from_bf16(const bf16* __restrict__ p) {
  v16bf f;
#pragma unroll
  for (int i = 0; i < 8; ++i) {
    f[i]     = p[i];
    f[i + 8] = p[i + 16];
  }
  return f;
}

// ---------------------------------------------------------------------------
// One-shot weight conversion fp32 -> bf16 (~1 MB total; removes all redundant
// per-wave weight conversions from the GEMM hot loops)
// ---------------------------------------------------------------------------
__global__ void cvt_f32_to_bf16(const float* __restrict__ src,
                                bf16* __restrict__ dst, int n) {
  const int i = blockIdx.x * blockDim.x + threadIdx.x;
  if (i < n) dst[i] = (__bf16)src[i];
}

// ---------------------------------------------------------------------------
// GEMM 1: z[BS,R] = x[BS,H] * in_basis[R,H]^T
// One wave computes a full 16x128 strip (8 tiles): x fragment converted once,
// reused for 8 independent wmma chains.
// ---------------------------------------------------------------------------
__global__ void __launch_bounds__(256) gemm_z(const float* __restrict__ x,
                                              const bf16* __restrict__ wib,
                                              bf16* __restrict__ z) {
  const int lane = threadIdx.x & 31;
  const int wave = threadIdx.x >> 5;
  const int tm = blockIdx.x * 8 + wave;           // 2048 row-tiles
  const int r = lane & 15, half = lane >> 4;
  v8f acc[8] = {};
  const float* arow = x + (size_t)(tm * 16 + r) * H_ + half * 8;
#pragma unroll 2
  for (int k0 = 0; k0 < H_; k0 += 32) {
    v16bf a = frag_from_f32(arow + k0);
#pragma unroll
    for (int t = 0; t < 8; ++t) {
      v16bf b = frag_from_bf16(wib + (size_t)(t * 16 + r) * H_ + k0 + half * 8);
      acc[t] = WMMA_BF16(a, b, acc[t]);
    }
  }
#pragma unroll
  for (int t = 0; t < 8; ++t)
#pragma unroll
    for (int v = 0; v < 8; ++v)
      z[(size_t)(tm * 16 + half * 8 + v) * R_ + t * 16 + r] = (__bf16)acc[t][v];
}

// ---------------------------------------------------------------------------
// GEMM 2 (fused): b,c,v = z * {k,q,v}_sel^T ; u = softplus(delta)*(b*v) ; c out
// One wave computes 4 n-tiles x 3 projections = 12 independent accumulators;
// z fragment reused 12x per k-step.
// ---------------------------------------------------------------------------
__global__ void __launch_bounds__(256) gemm_bcv(const bf16* __restrict__ z,
                                                const bf16* __restrict__ qsel,
                                                const bf16* __restrict__ ksel,
                                                const bf16* __restrict__ vsel,
                                                const float* __restrict__ delta_log,
                                                float* __restrict__ u,
                                                bf16* __restrict__ cgate) {
  const int lane = threadIdx.x & 31;
  const int wave = threadIdx.x >> 5;
  const int id = blockIdx.x * 8 + wave;           // 8192 waves
  const int tm = id >> 2;                         // row tile
  const int g  = id & 3;                          // group of 4 n-tiles
  const int r = lane & 15, half = lane >> 4;
  v8f accB[4] = {}, accC[4] = {}, accV[4] = {};
  const bf16* arow = z + (size_t)(tm * 16 + r) * R_ + half * 8;
#pragma unroll
  for (int k0 = 0; k0 < R_; k0 += 32) {           // 4 iterations
    v16bf a = frag_from_bf16(arow + k0);
#pragma unroll
    for (int t = 0; t < 4; ++t) {
      const size_t boff = (size_t)((g * 4 + t) * 16 + r) * R_ + k0 + half * 8;
      v16bf bb = frag_from_bf16(ksel + boff);
      v16bf bc = frag_from_bf16(qsel + boff);
      v16bf bv = frag_from_bf16(vsel + boff);
      accB[t] = WMMA_BF16(a, bb, accB[t]);
      accC[t] = WMMA_BF16(a, bc, accC[t]);
      accV[t] = WMMA_BF16(a, bv, accV[t]);
    }
  }
#pragma unroll
  for (int t = 0; t < 4; ++t) {
    const int col = (g * 4 + t) * 16 + r;
    const float dl = delta_log[col];
    const float delta = (dl > 20.f) ? dl : log1pf(expf(dl));   // softplus
#pragma unroll
    for (int v = 0; v < 8; ++v) {
      const size_t idx = (size_t)(tm * 16 + half * 8 + v) * N_ + col;
      u[idx]     = delta * accB[t][v] * accV[t][v];
      cgate[idx] = (__bf16)accC[t][v];
    }
  }
}

// ---------------------------------------------------------------------------
// Chunked diagonal scan: s_t = a*s_{t-1} + u_t ; y_t = c_t * s_t
// 32 chunks of 128 -> 65536-way parallel pass1/pass3, 2048-way carry pass.
// ---------------------------------------------------------------------------
__global__ void scan_pass1(const float* __restrict__ u,
                           const float* __restrict__ a_logit,
                           float* __restrict__ f) {
  const int n = threadIdx.x;              // 0..255
  const int chunk = blockIdx.x;           // 0..NCHUNK-1
  const int b = blockIdx.y;               // 0..B-1
  const float a = 1.f / (1.f + expf(-a_logit[n]));
  const float* up = u + ((size_t)b * S_ + (size_t)chunk * CSIZE) * N_ + n;
  float s = 0.f;
  for (int t = 0; t < CSIZE; ++t) s = fmaf(a, s, up[(size_t)t * N_]);
  f[((size_t)b * NCHUNK + chunk) * N_ + n] = s;
}

__global__ void scan_pass2(const float* __restrict__ f,
                           const float* __restrict__ state0,
                           const float* __restrict__ a_logit,
                           float* __restrict__ carry,
                           float* __restrict__ final_state) {
  const int n = threadIdx.x;
  const int b = blockIdx.x;
  const float a  = 1.f / (1.f + expf(-a_logit[n]));
  const float aL = powf(a, (float)CSIZE);
  float cy = state0[(size_t)b * N_ + n];
  for (int k = 0; k < NCHUNK; ++k) {
    const size_t idx = ((size_t)b * NCHUNK + k) * N_ + n;
    carry[idx] = cy;                          // exclusive carry into chunk k
    cy = fmaf(aL, cy, f[idx]);
  }
  final_state[(size_t)b * N_ + n] = cy;       // second reference output
}

__global__ void scan_pass3(const float* __restrict__ u,
                           const bf16* __restrict__ cgate,
                           const float* __restrict__ a_logit,
                           const float* __restrict__ carry,
                           bf16* __restrict__ y) {
  const int n = threadIdx.x;
  const int chunk = blockIdx.x;
  const int b = blockIdx.y;
  const float a = 1.f / (1.f + expf(-a_logit[n]));
  const size_t base = ((size_t)b * S_ + (size_t)chunk * CSIZE) * N_ + n;
  float s = carry[((size_t)b * NCHUNK + chunk) * N_ + n];
  for (int t = 0; t < CSIZE; ++t) {
    const size_t idx = base + (size_t)t * N_;
    s = fmaf(a, s, u[idx]);
    y[idx] = (__bf16)((float)cgate[idx] * s);
  }
}

// ---------------------------------------------------------------------------
// GEMM 3: out[BS,H] = y[BS,N] * out_proj[H,N]^T
// One wave computes a 16x128 strip (8 tiles); y fragment reused 8x.
// ---------------------------------------------------------------------------
__global__ void __launch_bounds__(256) gemm_out(const bf16* __restrict__ y,
                                                const bf16* __restrict__ wout,
                                                float* __restrict__ out) {
  const int lane = threadIdx.x & 31;
  const int wave = threadIdx.x >> 5;
  const int id = blockIdx.x * 8 + wave;           // 16384 waves
  const int tm = id >> 3;                         // row tile
  const int strip = id & 7;                       // 8 strips of 128 cols
  const int r = lane & 15, half = lane >> 4;
  v8f acc[8] = {};
  const bf16* arow = y + (size_t)(tm * 16 + r) * N_ + half * 8;
#pragma unroll
  for (int k0 = 0; k0 < N_; k0 += 32) {           // 8 iterations
    v16bf a = frag_from_bf16(arow + k0);
#pragma unroll
    for (int t = 0; t < 8; ++t) {
      const int nrow = (strip * 8 + t) * 16 + r;
      v16bf b = frag_from_bf16(wout + (size_t)nrow * N_ + k0 + half * 8);
      acc[t] = WMMA_BF16(a, b, acc[t]);
    }
  }
#pragma unroll
  for (int t = 0; t < 8; ++t)
#pragma unroll
    for (int v = 0; v < 8; ++v)
      out[(size_t)(tm * 16 + half * 8 + v) * H_ + strip * 128 + t * 16 + r] = acc[t][v];
}

// ---------------------------------------------------------------------------
extern "C" void kernel_launch(void* const* d_in, const int* in_sizes, int n_in,
                              void* d_out, int out_size, void* d_ws, size_t ws_size,
                              hipStream_t stream) {
  const float* x         = (const float*)d_in[0];
  const float* state0    = (const float*)d_in[1];
  const float* in_basis  = (const float*)d_in[2];
  const float* q_select  = (const float*)d_in[3];
  const float* k_select  = (const float*)d_in[4];
  const float* v_input   = (const float*)d_in[5];
  const float* out_proj  = (const float*)d_in[6];
  const float* a_logit   = (const float*)d_in[7];
  const float* delta_log = (const float*)d_in[8];

  float* out         = (float*)d_out;                 // [B,S,H]
  float* final_state = out + (size_t)B_ * S_ * H_;    // [B,N]

  // Workspace carve-out (~77 MB total)
  char* ws = (char*)d_ws;
  bf16*  z     = (bf16*)ws;  ws += sizeof(bf16)  * (size_t)BS_ * R_;        //  8.4 MB
  float* u     = (float*)ws; ws += sizeof(float) * (size_t)BS_ * N_;        // 33.6 MB
  bf16*  cgate = (bf16*)ws;  ws += sizeof(bf16)  * (size_t)BS_ * N_;        // 16.8 MB
  bf16*  y     = (bf16*)ws;  ws += sizeof(bf16)  * (size_t)BS_ * N_;        // 16.8 MB
  float* fch   = (float*)ws; ws += sizeof(float) * (size_t)B_ * NCHUNK * N_;
  float* carry = (float*)ws; ws += sizeof(float) * (size_t)B_ * NCHUNK * N_;
  bf16*  wib   = (bf16*)ws;  ws += sizeof(bf16)  * (size_t)R_ * H_;         // in_basis
  bf16*  wq    = (bf16*)ws;  ws += sizeof(bf16)  * (size_t)N_ * R_;         // q_select
  bf16*  wk    = (bf16*)ws;  ws += sizeof(bf16)  * (size_t)N_ * R_;         // k_select
  bf16*  wv    = (bf16*)ws;  ws += sizeof(bf16)  * (size_t)N_ * R_;         // v_input
  bf16*  wo    = (bf16*)ws;  ws += sizeof(bf16)  * (size_t)H_ * N_;         // out_proj

  // Weight conversion (tiny)
  cvt_f32_to_bf16<<<(R_ * H_ + 255) / 256, 256, 0, stream>>>(in_basis, wib, R_ * H_);
  cvt_f32_to_bf16<<<(N_ * R_ + 255) / 256, 256, 0, stream>>>(q_select, wq, N_ * R_);
  cvt_f32_to_bf16<<<(N_ * R_ + 255) / 256, 256, 0, stream>>>(k_select, wk, N_ * R_);
  cvt_f32_to_bf16<<<(N_ * R_ + 255) / 256, 256, 0, stream>>>(v_input,  wv, N_ * R_);
  cvt_f32_to_bf16<<<(H_ * N_ + 255) / 256, 256, 0, stream>>>(out_proj, wo, H_ * N_);

  gemm_z  <<<(BS_ / 16) / 8,           256, 0, stream>>>(x, wib, z);
  gemm_bcv<<<(BS_ / 16) * 4 / 8,       256, 0, stream>>>(z, wq, wk, wv, delta_log, u, cgate);
  scan_pass1<<<dim3(NCHUNK, B_), N_, 0, stream>>>(u, a_logit, fch);
  scan_pass2<<<B_, N_, 0, stream>>>(fch, state0, a_logit, carry, final_state);
  scan_pass3<<<dim3(NCHUNK, B_), N_, 0, stream>>>(u, cgate, a_logit, carry, y);
  gemm_out<<<(BS_ / 16) * 8 / 8,       256, 0, stream>>>(y, wo, out);
}
